// V8Layer_39900246180541
// MI455X (gfx1250) — matrix operand order repaired and checked by reading, not compiled, not run on hardware
//
#include <hip/hip_runtime.h>
#include <hip/hip_fp16.h>

// ---------------------------------------------------------------------------
// V8Layer for MI455X (gfx1250): f16 WMMA GEMMs (LDS-staged weights, 32x64 wave
// tiles) + fused causal scans + LayerNorm
// ---------------------------------------------------------------------------

typedef _Float16 v16h __attribute__((ext_vector_type(16)));
typedef _Float16 v8h  __attribute__((ext_vector_type(8)));
typedef float    v8f  __attribute__((ext_vector_type(8)));

#define B_ 4
#define T_ 2048
#define E_ 1024
#define D_ 4
#define M_ (B_*T_)        // 8192 tokens
#define H_ (E_/4)         // 256
#define LDSP 72           // padded LDS row stride in halves (36 words -> no bank conflicts)

__device__ __forceinline__ float sigmoidf_(float x){ return 1.0f/(1.0f + __expf(-x)); }
__device__ __forceinline__ float geluf_(float x){ return 0.5f*x*(1.0f + erff(x*0.70710678118654752f)); }
__device__ __forceinline__ float softplusf_(float x){ return log1pf(__expf(x)); }

// ---- WMMA fragment helpers (layouts per CDNA5 ISA 7.12.2, wave32) ----------
// A: 16x32 f16.  lane m = lane&15 ; halves 0..7 at k0+(lane>=16?8:0),
//                halves 8..15 at the same offset +16.
__device__ __forceinline__ v16h load_afrag(const _Float16* __restrict__ A, int lda,
                                           int row0, int k0, int lane){
  const _Float16* p = A + (size_t)(row0 + (lane & 15))*lda + k0 + ((lane >> 4) << 3);
  union { v16h v; v8h h[2]; } u;
  u.h[0] = *(const v8h*)(p);
  u.h[1] = *(const v8h*)(p + 16);
  return u.v;
}
// B fragment read out of the LDS-staged weight tile (row stride LDSP halves).
// B[k][n] = W[n][k]; lane n = lane&15 reads halves 0..7 of row n at
// kk + (lane>=16 ? 8 : 0), halves 8..15 at the same offset +16.  16B-granular reads (LDSP*2 is 16B aligned).
__device__ __forceinline__ v16h load_bfrag_lds(const _Float16* __restrict__ sh,
                                               int n, int kk, int lane){
  const _Float16* p = sh + (size_t)(n + (lane & 15))*LDSP + kk + ((lane >> 4) << 3);
  union { v16h v; v8h h[2]; } u;
  u.h[0] = *(const v8h*)(p);
  u.h[1] = *(const v8h*)(p + 16);
  return u.v;
}
__device__ __forceinline__ v8f wmma_f16(v16h a, v16h b, v8f c){
  return __builtin_amdgcn_wmma_f32_16x16x32_f16(false, a, false, b, (short)0, c, false, false);
}
// cooperative stage: W rows [nbase, nbase+nrows) x cols [k0, k0+64) -> sh
__device__ __forceinline__ void stage_w(const _Float16* __restrict__ W, int ldw,
                                        int nbase, int k0, _Float16* sh,
                                        int tid, int nrows){
  const int chunks = nrows*8;                   // 8 halves (16B) per chunk
  for (int c = tid; c < chunks; c += 256){
    int i = c >> 3, s = (c & 7)*8;
    *(v8h*)(sh + i*LDSP + s) = *(const v8h*)(W + (size_t)(nbase + i)*ldw + k0 + s);
  }
}

// ---- small utility kernels -------------------------------------------------
__global__ void k_f32_to_f16(const float* __restrict__ in, _Float16* __restrict__ out, int n){
  int i = blockIdx.x*blockDim.x + threadIdx.x;
  if (i < n) out[i] = (_Float16)in[i];
}

// softmax over D of blend (D,E) -> bw (D,E)
__global__ void k_blendw(const float* __restrict__ blend, float* __restrict__ bw){
  int e = blockIdx.x*blockDim.x + threadIdx.x;
  if (e >= E_) return;
  float v[D_]; float mx = -3.0e38f;
  #pragma unroll
  for (int d=0; d<D_; ++d){ v[d] = blend[d*E_ + e]; mx = fmaxf(mx, v[d]); }
  float s = 0.f;
  #pragma unroll
  for (int d=0; d<D_; ++d){ v[d] = __expf(v[d]-mx); s += v[d]; }
  float inv = 1.0f/s;
  #pragma unroll
  for (int d=0; d<D_; ++d) bw[d*E_ + e] = v[d]*inv;
}

// causal EMA over T for ctx: one thread per (b,e) channel
__global__ void k_ema_ctx(const float* __restrict__ x, const float* __restrict__ tau,
                          _Float16* __restrict__ ctx){
  int tid = blockIdx.x*blockDim.x + threadIdx.x;     // 0..B*E-1
  int b = tid >> 10, e = tid & (E_-1);
  float alpha = __expf(-softplusf_(tau[0]));
  const float* xp = x + (size_t)b*T_*E_ + e;
  _Float16*   cp = ctx + (size_t)b*T_*E_ + e;
  float h = 0.f;
  for (int t=0; t<T_; ++t){
    h = alpha*h + xp[(size_t)t*E_];
    cp[(size_t)t*E_] = (_Float16)h;
  }
}

// fused 4-dendrite EMA of (gated*mod) + blend -> temporal (f16)
__global__ void k_ema_temporal(const _Float16* __restrict__ gated, const float* __restrict__ mod,
                               const float* __restrict__ taus, const float* __restrict__ bw,
                               _Float16* __restrict__ temporal){
  int tid = blockIdx.x*blockDim.x + threadIdx.x;     // 0..B*E-1
  int b = tid >> 10, e = tid & (E_-1);
  float al[D_], w[D_], h[D_];
  #pragma unroll
  for (int d=0; d<D_; ++d){
    al[d] = __expf(-softplusf_(taus[d]));
    w[d]  = bw[d*E_ + e];
    h[d]  = 0.f;
  }
  const _Float16* gp = gated + (size_t)b*T_*E_ + e;
  const float4*   mp = (const float4*)(mod + (size_t)b*T_*D_);
  _Float16*       tp = temporal + (size_t)b*T_*E_ + e;
  for (int t=0; t<T_; ++t){
    float g = (float)gp[(size_t)t*E_];
    float4 m = mp[t];
    h[0] = al[0]*h[0] + g*m.x;
    h[1] = al[1]*h[1] + g*m.y;
    h[2] = al[2]*h[2] + g*m.z;
    h[3] = al[3]*h[3] + g*m.w;
    float s = w[0]*h[0] + w[1]*h[1] + w[2]*h[2] + w[3]*h[3];
    tp[(size_t)t*E_] = (_Float16)s;
  }
}

// mod = 0.5 + sigmoid(h1 @ Wn2^T + bn2), N = D = 4 (tiny, plain dot)
__global__ void k_mod(const _Float16* __restrict__ h1, const float* __restrict__ Wn2,
                      const float* __restrict__ bn2, float* __restrict__ mod){
  int gid = blockIdx.x*blockDim.x + threadIdx.x;
  if (gid >= M_*D_) return;
  int tok = gid >> 2, d = gid & 3;
  const _Float16* hp = h1 + (size_t)tok*H_;
  const float*    wp = Wn2 + d*H_;
  float acc = bn2[d];
  for (int i=0; i<H_; ++i) acc += (float)hp[i]*wp[i];
  mod[gid] = 0.5f + sigmoidf_(acc);
}

// ---- WMMA GEMM kernels -----------------------------------------------------
// Single-weight kernels: block = 256 thr = 8 waves (4 M x 2 N), block tile
// 128x128, wave tile 32x64.  Weight K-slab (128 rows x 64 halves) staged in
// LDS once per block and shared by the 4 M-waves of each N column.

// gated = x * sigmoid([x|ctx] @ Wg^T + bg)   (K = 2E via two phases)
__global__ void k_gemm_gate(const _Float16* __restrict__ Xh, const _Float16* __restrict__ Ch,
                            const _Float16* __restrict__ Wgh, const float* __restrict__ xf,
                            const float* __restrict__ bg, _Float16* __restrict__ gated){
  __shared__ _Float16 sh[128*LDSP];
  const int tid = threadIdx.x, lane = tid & 31, wave = tid >> 5;
  const int mw = wave & 3, nw = wave >> 2;
  const int row0 = blockIdx.x*128 + mw*32;
  const int nblk = blockIdx.y*128;
  v8f acc0[4] = {}, acc1[4] = {};
  #pragma unroll 1
  for (int phase = 0; phase < 2; ++phase){
    const _Float16* A = phase ? Ch : Xh;
    const int kbase = phase*E_;
    for (int k0 = 0; k0 < E_; k0 += 64){
      __syncthreads();
      stage_w(Wgh, 2*E_, nblk, kbase + k0, sh, tid, 128);
      __syncthreads();
      #pragma unroll
      for (int kk = 0; kk < 64; kk += 32){
        v16h a0 = load_afrag(A, E_, row0,      k0 + kk, lane);
        v16h a1 = load_afrag(A, E_, row0 + 16, k0 + kk, lane);
        #pragma unroll
        for (int j = 0; j < 4; ++j){
          v16h b = load_bfrag_lds(sh, nw*64 + j*16, kk, lane);
          acc0[j] = wmma_f16(a0, b, acc0[j]);
          acc1[j] = wmma_f16(a1, b, acc1[j]);
        }
      }
    }
  }
  #pragma unroll
  for (int j = 0; j < 4; ++j){
    int n = nblk + nw*64 + j*16 + (lane & 15);
    float bgn = bg[n];
    #pragma unroll
    for (int r = 0; r < 8; ++r){
      int m = row0 + r + ((lane >= 16) ? 8 : 0);
      size_t i0 = (size_t)m*E_ + n;
      gated[i0] = (_Float16)(xf[i0]*sigmoidf_(acc0[j][r] + bgn));
      size_t i1 = i0 + (size_t)16*E_;
      gated[i1] = (_Float16)(xf[i1]*sigmoidf_(acc1[j][r] + bgn));
    }
  }
}

// h1 = gelu(x @ Wn1^T + bn1), N = 256
__global__ void k_gemm_h1(const _Float16* __restrict__ Xh, const _Float16* __restrict__ Wh,
                          const float* __restrict__ bn1, _Float16* __restrict__ h1){
  __shared__ _Float16 sh[128*LDSP];
  const int tid = threadIdx.x, lane = tid & 31, wave = tid >> 5;
  const int mw = wave & 3, nw = wave >> 2;
  const int row0 = blockIdx.x*128 + mw*32;
  const int nblk = blockIdx.y*128;
  v8f acc0[4] = {}, acc1[4] = {};
  for (int k0 = 0; k0 < E_; k0 += 64){
    __syncthreads();
    stage_w(Wh, E_, nblk, k0, sh, tid, 128);
    __syncthreads();
    #pragma unroll
    for (int kk = 0; kk < 64; kk += 32){
      v16h a0 = load_afrag(Xh, E_, row0,      k0 + kk, lane);
      v16h a1 = load_afrag(Xh, E_, row0 + 16, k0 + kk, lane);
      #pragma unroll
      for (int j = 0; j < 4; ++j){
        v16h b = load_bfrag_lds(sh, nw*64 + j*16, kk, lane);
        acc0[j] = wmma_f16(a0, b, acc0[j]);
        acc1[j] = wmma_f16(a1, b, acc1[j]);
      }
    }
  }
  #pragma unroll
  for (int j = 0; j < 4; ++j){
    int n = nblk + nw*64 + j*16 + (lane & 15);
    float bn = bn1[n];
    #pragma unroll
    for (int r = 0; r < 8; ++r){
      int m = row0 + r + ((lane >= 16) ? 8 : 0);
      h1[(size_t)m*H_ + n]        = (_Float16)geluf_(acc0[j][r] + bn);
      h1[(size_t)(m+16)*H_ + n]   = (_Float16)geluf_(acc1[j][r] + bn);
    }
  }
}

// outp = sg @ Wout^T (f32 store, no bias)
__global__ void k_gemm_out(const _Float16* __restrict__ Ah, const _Float16* __restrict__ Wh,
                           float* __restrict__ outp){
  __shared__ _Float16 sh[128*LDSP];
  const int tid = threadIdx.x, lane = tid & 31, wave = tid >> 5;
  const int mw = wave & 3, nw = wave >> 2;
  const int row0 = blockIdx.x*128 + mw*32;
  const int nblk = blockIdx.y*128;
  v8f acc0[4] = {}, acc1[4] = {};
  for (int k0 = 0; k0 < E_; k0 += 64){
    __syncthreads();
    stage_w(Wh, E_, nblk, k0, sh, tid, 128);
    __syncthreads();
    #pragma unroll
    for (int kk = 0; kk < 64; kk += 32){
      v16h a0 = load_afrag(Ah, E_, row0,      k0 + kk, lane);
      v16h a1 = load_afrag(Ah, E_, row0 + 16, k0 + kk, lane);
      #pragma unroll
      for (int j = 0; j < 4; ++j){
        v16h b = load_bfrag_lds(sh, nw*64 + j*16, kk, lane);
        acc0[j] = wmma_f16(a0, b, acc0[j]);
        acc1[j] = wmma_f16(a1, b, acc1[j]);
      }
    }
  }
  #pragma unroll
  for (int j = 0; j < 4; ++j){
    int n = nblk + nw*64 + j*16 + (lane & 15);
    #pragma unroll
    for (int r = 0; r < 8; ++r){
      int m = row0 + r + ((lane >= 16) ? 8 : 0);
      outp[(size_t)m*E_ + n]      = acc0[j][r];
      outp[(size_t)(m+16)*E_ + n] = acc1[j][r];
    }
  }
}

// sg = gelu(x@Wsoma^T+bsoma) * sigmoid(temporal@Wtemp^T+btemp): fused dual GEMM.
// 8 waves stacked in M (block tile 128x64, wave tile 16x64); both weight slabs
// (64 rows x 64 halves each) staged in LDS.
__global__ void k_gemm_gs(const _Float16* __restrict__ Th, const _Float16* __restrict__ Xh,
                          const _Float16* __restrict__ Wth, const _Float16* __restrict__ Wsh,
                          const float* __restrict__ btemp, const float* __restrict__ bsoma,
                          _Float16* __restrict__ sg){
  __shared__ _Float16 shT[64*LDSP];
  __shared__ _Float16 shS[64*LDSP];
  const int tid = threadIdx.x, lane = tid & 31, wave = tid >> 5;
  const int row0 = blockIdx.x*128 + wave*16;
  const int n0   = blockIdx.y*64;
  v8f accG[4] = {}, accS[4] = {};
  for (int k0 = 0; k0 < E_; k0 += 64){
    __syncthreads();
    stage_w(Wth, E_, n0, k0, shT, tid, 64);
    stage_w(Wsh, E_, n0, k0, shS, tid, 64);
    __syncthreads();
    #pragma unroll
    for (int kk = 0; kk < 64; kk += 32){
      v16h a1 = load_afrag(Th, E_, row0, k0 + kk, lane);
      v16h a2 = load_afrag(Xh, E_, row0, k0 + kk, lane);
      #pragma unroll
      for (int j = 0; j < 4; ++j){
        accG[j] = wmma_f16(a1, load_bfrag_lds(shT, j*16, kk, lane), accG[j]);
        accS[j] = wmma_f16(a2, load_bfrag_lds(shS, j*16, kk, lane), accS[j]);
      }
    }
  }
  #pragma unroll
  for (int j = 0; j < 4; ++j){
    int n = n0 + j*16 + (lane & 15);
    float bt = btemp[n], bs = bsoma[n];
    #pragma unroll
    for (int r = 0; r < 8; ++r){
      int m = row0 + r + ((lane >= 16) ? 8 : 0);
      float g = sigmoidf_(accG[j][r] + bt);
      float s = geluf_(accS[j][r] + bs);
      sg[(size_t)m*E_ + n] = (_Float16)(s*g);
    }
  }
}

// row LayerNorm over E=1024 (one block / token row)
__global__ void k_layernorm(const float* __restrict__ outp, const float* __restrict__ gamma,
                            const float* __restrict__ beta, float* __restrict__ y){
  __shared__ float s1[256], s2[256];
  const int row = blockIdx.x, tid = threadIdx.x;
  const float* p = outp + (size_t)row*E_;
  float v[4], a = 0.f, b = 0.f;
  #pragma unroll
  for (int i=0; i<4; ++i){ v[i] = p[tid + 256*i]; a += v[i]; b += v[i]*v[i]; }
  s1[tid] = a; s2[tid] = b;
  __syncthreads();
  for (int off=128; off>0; off >>= 1){
    if (tid < off){ s1[tid] += s1[tid+off]; s2[tid] += s2[tid+off]; }
    __syncthreads();
  }
  float mu   = s1[0]*(1.0f/E_);
  float var  = s2[0]*(1.0f/E_) - mu*mu;
  float rstd = rsqrtf(var + 1e-5f);
  #pragma unroll
  for (int i=0; i<4; ++i){
    int c = tid + 256*i;
    y[(size_t)row*E_ + c] = (v[i]-mu)*rstd*gamma[c] + beta[c];
  }
}

// ---------------------------------------------------------------------------
extern "C" void kernel_launch(void* const* d_in, const int* in_sizes, int n_in,
                              void* d_out, int out_size, void* d_ws, size_t ws_size,
                              hipStream_t stream) {
  (void)in_sizes; (void)n_in; (void)out_size; (void)ws_size;
  const float* x       = (const float*)d_in[0];
  const float* tau_ctx = (const float*)d_in[1];
  const float* tau_raw = (const float*)d_in[2];
  const float* Wg      = (const float*)d_in[3];
  const float* bg      = (const float*)d_in[4];
  const float* Wn1     = (const float*)d_in[5];
  const float* bn1     = (const float*)d_in[6];
  const float* Wn2     = (const float*)d_in[7];
  const float* bn2     = (const float*)d_in[8];
  const float* blend   = (const float*)d_in[9];
  const float* Wsoma   = (const float*)d_in[10];
  const float* bsoma   = (const float*)d_in[11];
  const float* Wtemp   = (const float*)d_in[12];
  const float* btemp   = (const float*)d_in[13];
  const float* Wout    = (const float*)d_in[14];
  const float* gamma   = (const float*)d_in[15];
  const float* beta    = (const float*)d_in[16];
  float* y = (float*)d_out;

  // workspace carve-up (256B aligned)
  char* ws = (char*)d_ws;
  size_t off = 0;
  auto carve = [&](size_t bytes)->void* {
    void* p = ws + off;
    off = (off + bytes + 255) & ~(size_t)255;
    return p;
  };
  _Float16* xh    = (_Float16*)carve((size_t)M_*E_*2);   // x in f16
  _Float16* ch    = (_Float16*)carve((size_t)M_*E_*2);   // ctx f16
  _Float16* gth   = (_Float16*)carve((size_t)M_*E_*2);   // gated f16
  _Float16* tmh   = (_Float16*)carve((size_t)M_*E_*2);   // temporal f16
  _Float16* sgh   = (_Float16*)carve((size_t)M_*E_*2);   // soma*gate f16
  _Float16* h1h   = (_Float16*)carve((size_t)M_*H_*2);   // h1 f16
  float*    modb  = (float*)   carve((size_t)M_*D_*4);   // mod f32
  float*    outp  = (float*)   carve((size_t)M_*E_*4);   // pre-LN out f32
  _Float16* Wgh   = (_Float16*)carve((size_t)E_*2*E_*2); // Wg f16
  _Float16* Wn1h  = (_Float16*)carve((size_t)H_*E_*2);   // Wn1 f16
  _Float16* Wth   = (_Float16*)carve((size_t)E_*E_*2);   // Wtemp f16
  _Float16* Wsh   = (_Float16*)carve((size_t)E_*E_*2);   // Wsoma f16
  _Float16* Woh   = (_Float16*)carve((size_t)E_*E_*2);   // Wout f16
  float*    bw    = (float*)   carve((size_t)D_*E_*4);   // softmax(blend)

  const int TPB = 256;
  auto cvt = [&](const float* src, _Float16* dst, int n){
    k_f32_to_f16<<<(n + TPB - 1)/TPB, TPB, 0, stream>>>(src, dst, n);
  };
  // precision converts
  cvt(x,     xh,   M_*E_);
  cvt(Wg,    Wgh,  E_*2*E_);
  cvt(Wn1,   Wn1h, H_*E_);
  cvt(Wtemp, Wth,  E_*E_);
  cvt(Wsoma, Wsh,  E_*E_);
  cvt(Wout,  Woh,  E_*E_);

  // blend softmax + ctx EMA
  k_blendw<<<(E_ + TPB - 1)/TPB, TPB, 0, stream>>>(blend, bw);
  k_ema_ctx<<<(B_*E_)/TPB, TPB, 0, stream>>>(x, tau_ctx, ch);

  dim3 blk(TPB);
  dim3 gE(M_/128, E_/128);   // 64 x 8  (block tile 128x128)
  dim3 gH(M_/128, H_/128);   // 64 x 2
  dim3 gGS(M_/128, E_/64);   // 64 x 16 (block tile 128x64)

  // gated = x * sigmoid([x|ctx] Wg^T + bg)
  k_gemm_gate<<<gE, blk, 0, stream>>>(xh, ch, Wgh, x, bg, gth);
  // h1 = gelu(x Wn1^T + bn1)
  k_gemm_h1<<<gH, blk, 0, stream>>>(xh, Wn1h, bn1, h1h);
  // mod = 0.5 + sigmoid(h1 Wn2^T + bn2)
  k_mod<<<(M_*D_)/TPB, TPB, 0, stream>>>(h1h, Wn2, bn2, modb);
  // temporal = sum_d softmax(blend)_d * EMA_d(gated * mod_d)
  k_ema_temporal<<<(B_*E_)/TPB, TPB, 0, stream>>>(gth, modb, tau_raw, bw, tmh);
  // sg = gelu(x Wsoma^T + bsoma) * sigmoid(temporal Wtemp^T + btemp)
  k_gemm_gs<<<gGS, blk, 0, stream>>>(tmh, xh, Wth, Wsh, btemp, bsoma, sgh);
  // outp = sg Wout^T
  k_gemm_out<<<gE, blk, 0, stream>>>(sgh, Woh, outp);
  // y = layernorm(outp) * gamma + beta
  k_layernorm<<<M_, blk, 0, stream>>>(outp, gamma, beta, y);
}
